// KAN_16389595202321
// MI455X (gfx1250) — compile-verified, matrix operand order checked by reading, not run on hardware
//
#include <hip/hip_runtime.h>
#include <hip/hip_bf16.h>

typedef __attribute__((ext_vector_type(16))) _Float16 v16h;
typedef __attribute__((ext_vector_type(8)))  _Float16 v8h;
typedef __attribute__((ext_vector_type(4)))  _Float16 v4h;
typedef __attribute__((ext_vector_type(8)))  float    v8f;

#define IN_DIM    32
#define BASIS     11      // real basis funcs per feature
#define KDIM      384     // padded contraction length (32 feat * 12)
#define HIDDEN    64
#define KSTEPS    (KDIM / 32)      // 12 wmma K-steps
#define WAVES     2                // waves per block (2 M-tiles per wave)
#define ROW_H     392              // basis LDS row stride in halves (784 B)

// ---------------------------------------------------------------------------
// Prep kernel: W [32][11][64] f32  ->  Wt f16 [64][384] (transposed + padded),
// and biasSum[64] = sum_f b[f][:]
// k-mapping (must match basis packing in main kernel):
//   region A: k = f*8 + c           for c in 0..7
//   region B: k = 256 + f*4 + (c-8) for c in 8..11 (c==11 is the zero pad)
// ---------------------------------------------------------------------------
__global__ void kan_prep(const float* __restrict__ W, const float* __restrict__ b,
                         _Float16* __restrict__ wtg, float* __restrict__ biasSum) {
    int tid = blockIdx.x * blockDim.x + threadIdx.x;
    int nthreads = gridDim.x * blockDim.x;
    for (int idx = tid; idx < HIDDEN * KDIM; idx += nthreads) {
        int n = idx / KDIM;
        int k = idx % KDIM;
        float val;
        if (k < 256) {
            int f = k >> 3, c = k & 7;
            val = W[(f * BASIS + c) * HIDDEN + n];
        } else {
            int k2 = k - 256;
            int f = k2 >> 2, c = 8 + (k2 & 3);
            val = (c < BASIS) ? W[(f * BASIS + c) * HIDDEN + n] : 0.0f;
        }
        wtg[n * KDIM + k] = (_Float16)val;
    }
    for (int n = tid; n < HIDDEN; n += nthreads) {
        float s = 0.0f;
        for (int f = 0; f < IN_DIM; ++f) s += b[f * HIDDEN + n];
        biasSum[n] = s;
    }
}

// ---------------------------------------------------------------------------
// Main kernel: each wave computes TWO 16-token x 64-hidden tiles (32 tokens),
// sharing B fragments between the tiles: 96 v_wmma per wave vs 48 B-frag
// load pairs. Harmonics 2..5 via angle-addition recurrence (2 trans/lane/token
// instead of 10) since v_sin/v_cos are the VALU-side bottleneck while WMMA
// co-executes.
// ---------------------------------------------------------------------------
__global__ __launch_bounds__(WAVES * 32) void kan_wmma(
        const float* __restrict__ x, const _Float16* __restrict__ wtg,
        const float* __restrict__ biasSum, float* __restrict__ out, int numTok) {

    __shared__ _Float16 basis[WAVES * 32 * ROW_H];   // 50176 bytes

    const int tid  = threadIdx.x;
    const int wave = tid >> 5;
    const int lane = tid & 31;
    const int lo   = lane & 15;        // N column / M row within half-wave
    const int hi   = lane >> 4;        // selects K sub-block per 16-bit WMMA layout
    const int tok0 = (blockIdx.x * WAVES + wave) * 32;
    if (tok0 >= numTok) return;        // wave-uniform: EXEC stays all-ones below

    // ---- Fourier basis for 32 tokens; lane == feature index ---------------
    _Float16* bw = &basis[wave * 32 * ROW_H];
#pragma unroll
    for (int t = 0; t < 32; ++t) {
        int tt = tok0 + t;
        if (tt >= numTok) tt = numTok - 1;          // clamp (full tiles in practice)
        float a = x[tt * IN_DIM + lane];

        float s1 = __sinf(a), c1 = __cosf(a);
        // angle-addition recurrence: sin/cos(k*a) from sin/cos((k-1)*a)
        float s2 = s1 * c1 + c1 * s1;
        float c2 = c1 * c1 - s1 * s1;
        float s3 = s2 * c1 + c2 * s1;
        float c3 = c2 * c1 - s2 * s1;
        float s4 = s3 * c1 + c3 * s1;
        float c4 = c3 * c1 - s3 * s1;
        float s5 = s4 * c1 + c4 * s1;
        float c5 = c4 * c1 - s4 * s1;

        v8h pa;                                    // c = 0..7
        pa[0] = (_Float16)1.0f; pa[1] = (_Float16)s1; pa[2] = (_Float16)c1;
        pa[3] = (_Float16)s2;   pa[4] = (_Float16)c2; pa[5] = (_Float16)s3;
        pa[6] = (_Float16)c3;   pa[7] = (_Float16)s4;
        v4h pb;                                    // c = 8..11 (pad 0)
        pb[0] = (_Float16)c4;   pb[1] = (_Float16)s5;
        pb[2] = (_Float16)c5;   pb[3] = (_Float16)0.0f;
        // region A: k = lane*8 + c   (16-B aligned -> ds_store_b128)
        *(v8h*)(bw + t * ROW_H + lane * 8) = pa;
        // region B: k = 256 + lane*4 + (c-8)  (8-B aligned -> ds_store_b64)
        *(v4h*)(bw + t * ROW_H + 256 + lane * 4) = pb;
    }
    // Same-wave DS store -> DS load handoff; CDNA5 split counter fence.
    asm volatile("s_wait_dscnt 0" ::: "memory");

    // ---- WMMA main loop: 2 M-tiles share each B fragment -------------------
    v8f accA0 = {}, accA1 = {}, accA2 = {}, accA3 = {};   // tile 0 (tokens 0..15)
    v8f accB0 = {}, accB1 = {}, accB2 = {}, accB3 = {};   // tile 1 (tokens 16..31)
    const _Float16* arow0 = bw + lo * ROW_H;              // A row, tile 0
    const _Float16* arow1 = bw + (16 + lo) * ROW_H;       // A row, tile 1
    const _Float16* brow  = wtg + lo * KDIM;              // B column base
#pragma unroll
    for (int kb = 0; kb < KSTEPS; ++kb) {
        // A fragments: lane(lo,hi) holds K = kb*32 + hi*8 + {0..7, 16..23}
        v8h a00 = *(const v8h*)(arow0 + kb * 32 + hi * 8);
        v8h a01 = *(const v8h*)(arow0 + kb * 32 + hi * 8 + 16);
        v16h af0 = __builtin_shufflevector(a00, a01, 0,1,2,3,4,5,6,7,8,9,10,11,12,13,14,15);
        v8h a10 = *(const v8h*)(arow1 + kb * 32 + hi * 8);
        v8h a11 = *(const v8h*)(arow1 + kb * 32 + hi * 8 + 16);
        v16h af1 = __builtin_shufflevector(a10, a11, 0,1,2,3,4,5,6,7,8,9,10,11,12,13,14,15);

#pragma unroll
        for (int nt = 0; nt < 4; ++nt) {
            const _Float16* bp = brow + nt * 16 * KDIM + kb * 32 + hi * 8;
            v8h b0 = *(const v8h*)(bp);
            v8h b1 = *(const v8h*)(bp + 16);
            v16h bf = __builtin_shufflevector(b0, b1, 0,1,2,3,4,5,6,7,8,9,10,11,12,13,14,15);
            switch (nt) {
              case 0:
                accA0 = __builtin_amdgcn_wmma_f32_16x16x32_f16(false, af0, false, bf, (short)0, accA0, false, false);
                accB0 = __builtin_amdgcn_wmma_f32_16x16x32_f16(false, af1, false, bf, (short)0, accB0, false, false);
                break;
              case 1:
                accA1 = __builtin_amdgcn_wmma_f32_16x16x32_f16(false, af0, false, bf, (short)0, accA1, false, false);
                accB1 = __builtin_amdgcn_wmma_f32_16x16x32_f16(false, af1, false, bf, (short)0, accB1, false, false);
                break;
              case 2:
                accA2 = __builtin_amdgcn_wmma_f32_16x16x32_f16(false, af0, false, bf, (short)0, accA2, false, false);
                accB2 = __builtin_amdgcn_wmma_f32_16x16x32_f16(false, af1, false, bf, (short)0, accB2, false, false);
                break;
              case 3:
                accA3 = __builtin_amdgcn_wmma_f32_16x16x32_f16(false, af0, false, bf, (short)0, accA3, false, false);
                accB3 = __builtin_amdgcn_wmma_f32_16x16x32_f16(false, af1, false, bf, (short)0, accB3, false, false);
                break;
            }
        }
    }

    // ---- bias + store (C/D layout: VGPR r -> M = r + 8*hi, N = lo) ---------
    float b0 = biasSum[lo];
    float b1 = biasSum[16 + lo];
    float b2 = biasSum[32 + lo];
    float b3 = biasSum[48 + lo];
#pragma unroll
    for (int r = 0; r < 8; ++r) {
        int trow = tok0 + r + hi * 8;
        if (trow < numTok) {
            float* orow = out + trow * HIDDEN + lo;
            orow[0]  = accA0[r] + b0;
            orow[16] = accA1[r] + b1;
            orow[32] = accA2[r] + b2;
            orow[48] = accA3[r] + b3;
        }
        int trow1 = tok0 + 16 + r + hi * 8;
        if (trow1 < numTok) {
            float* orow = out + trow1 * HIDDEN + lo;
            orow[0]  = accB0[r] + b0;
            orow[16] = accB1[r] + b1;
            orow[32] = accB2[r] + b2;
            orow[48] = accB3[r] + b3;
        }
    }
}

// ---------------------------------------------------------------------------
extern "C" void kernel_launch(void* const* d_in, const int* in_sizes, int n_in,
                              void* d_out, int out_size, void* d_ws, size_t ws_size,
                              hipStream_t stream) {
    const float* x = (const float*)d_in[0];     // [B,T,32] f32
    const float* W = (const float*)d_in[1];     // [32,11,64] f32
    const float* b = (const float*)d_in[2];     // [32,64] f32
    float* out = (float*)d_out;                 // [B,T,64] f32

    _Float16* wtg   = (_Float16*)d_ws;                          // 64*384 f16 = 48 KB
    float* biasSum  = (float*)((char*)d_ws + HIDDEN * KDIM * sizeof(_Float16));

    int numTok = in_sizes[0] / IN_DIM;          // B*T

    kan_prep<<<96, 256, 0, stream>>>(W, b, wtg, biasSum);

    int tokPerBlock = WAVES * 32;
    int blocks = (numTok + tokPerBlock - 1) / tokPerBlock;
    kan_wmma<<<blocks, WAVES * 32, 0, stream>>>(x, wtg, biasSum, out, numTok);
}
